// LocalMessagePassing_4501125726901
// MI455X (gfx1250) — matrix operand order repaired and checked by reading, not compiled, not run on hardware
//
#include <hip/hip_runtime.h>
#include <hip/hip_bf16.h>
#include <math.h>

// ---------------------------------------------------------------------------
// LocalMessagePassing on MI455X (gfx1250, wave32, WMMA 16x16x32 f16)
//
// Algebraic optimization: concat([self,nbr]) @ W1 == self@W1s + nbr@W1n,
// so per step we run 6 GEMMs of [32768x256]x[256x256] on the f16 WMMA path
// (f32 accumulate), instead of 5x-redundant 512-K GEMMs.
// ---------------------------------------------------------------------------

typedef __attribute__((ext_vector_type(16))) _Float16 v16h;
typedef __attribute__((ext_vector_type(8)))  _Float16 v8h;
typedef __attribute__((ext_vector_type(8)))  float    v8f;

#define DIM 256
#define BATCH 4
#define SEQ 8192
#define ROWS_TOTAL (BATCH*SEQ)        // 32768
#define TILE_M 32
#define NUM_WG (ROWS_TOTAL/TILE_M)    // 1024
#define LDH 264                       // padded f16 LDS row stride (bank rotate)
#define LDF 260                       // padded f32 LDS row stride

__device__ __forceinline__ float gelu_exact(float x) {
  return 0.5f * x * (1.0f + erff(x * 0.70710678118654752f));
}

// WMMA D = A*B + C, f16 in / f32 acc
__device__ __forceinline__ v8f wmma_f16(v16h a, v16h b, v8f c) {
  return __builtin_amdgcn_wmma_f32_16x16x32_f16(false, a, false, b, (short)0, c,
                                                false, false);
}

// Load a 16x32 f16 A fragment from a row-major LDS tile (stride LDH).
// Lane l (<16): M=l, K=0..7 (lo) and K=16..23 (hi); lane>=16: M=l-16,
// K=8..15 / K=24..31  -> two 16B LDS loads per lane.
__device__ __forceinline__ v16h load_a_lds(const _Float16* base, int lane) {
  int m  = lane & 15;
  int kb = (lane >> 4) << 3;
  const _Float16* p = base + m * LDH + kb;
  v8h lo = *(const v8h*)(p);
  v8h hi = *(const v8h*)(p + 16);
  return __builtin_shufflevector(lo, hi, 0, 1, 2, 3, 4, 5, 6, 7, 8, 9, 10, 11,
                                 12, 13, 14, 15);
}

// Load a pre-packed 32x16 B fragment: one contiguous 32B load per lane.
__device__ __forceinline__ v16h load_b_pk(const _Float16* Wpk, int kt, int nt,
                                          int lane) {
  return *(const v16h*)(Wpk + ((((kt << 4) + nt) << 5) + lane) * 16);
}

// Copy 32 rows x 256 f16 (global, stride 256) -> LDS (stride LDH).
__device__ __forceinline__ void stage_rows(_Float16* lds, const _Float16* g,
                                           int tid) {
  int row = tid >> 3, ch = tid & 7;     // 8 threads/row, 4x uint4 each
  const uint4* src = (const uint4*)(g + (size_t)row * DIM) + ch * 4;
  uint4* dst = (uint4*)(lds + row * LDH) + ch * 4;
  dst[0] = src[0]; dst[1] = src[1]; dst[2] = src[2]; dst[3] = src[3];
}

// ------------------------- weight pre-pack ---------------------------------
// Pack row-major f32 W[k][n] into WMMA B-fragment order, f16:
// idx = ((kt*16+nt)*32 + lane)*16 + i ;  lane<16: n=nt*16+lane, k=kt*32+(i<8 ? i : 8+i)
//                                        lane>=16: n=nt*16+lane-16, k=kt*32+8+(i<8 ? i : 8+i)
__global__ __launch_bounds__(256) void k_pack(const float* __restrict__ W,
                                              _Float16* __restrict__ dst,
                                              int row_off) {
  int pidx = blockIdx.x * 256 + threadIdx.x;  // < 65536
  int i = pidx & 15;
  int lane = (pidx >> 4) & 31;
  int nt = (pidx >> 9) & 15;
  int kt = pidx >> 13;
  int n = nt * 16 + (lane & 15);
  int k = kt * 32 + ((lane >> 4) << 3) + (i & 7) + ((i >> 3) << 4);
  dst[pidx] = (_Float16)W[(size_t)(row_off + k) * DIM + n];
}

// ------------------------------ init ---------------------------------------
__global__ __launch_bounds__(256) void k_init(const float* __restrict__ in,
                                              float* __restrict__ h32,
                                              _Float16* __restrict__ h16,
                                              float* __restrict__ out,
                                              int* flags, float* remaining,
                                              float* kl) {
  size_t i = (size_t)blockIdx.x * 256 + threadIdx.x;
  if (i < (size_t)ROWS_TOTAL * DIM) {
    float v = in[i];
    h32[i] = v;
    h16[i] = (_Float16)v;
    out[i] = 0.f;
  }
  if (i == 0) {
    out[(size_t)ROWS_TOTAL * DIM] = 0.f;  // kl slot
    flags[0] = 1;
    kl[0] = 0.f;
    remaining[0] = remaining[1] = remaining[2] = remaining[3] = 1.f;
  }
}

// --------------------------- K1: S,T projection ----------------------------
// S = h@W1s + b1 ; T = h@W1n   (both f16 out). 8 waves: (mt in 0..1, nq 0..3).
__global__ __launch_bounds__(256) void k_proj(
    const _Float16* __restrict__ h16, const _Float16* __restrict__ W1s,
    const _Float16* __restrict__ W1n, const float* __restrict__ b1,
    _Float16* __restrict__ S, _Float16* __restrict__ T,
    const int* __restrict__ flags) {
  if (!flags[0]) return;
  __shared__ __align__(16) _Float16 hA[32 * LDH];
  int tid = threadIdx.x;
  int R = blockIdx.x * TILE_M;
  stage_rows(hA, h16 + (size_t)R * DIM, tid);
  __syncthreads();
  int lane = tid & 31, wv = tid >> 5;
  int mt = wv >> 2, nq = wv & 3;
  v8f aS[4] = {};
  v8f aT[4] = {};
  const _Float16* abase = hA + mt * 16 * LDH;
#pragma unroll
  for (int kt = 0; kt < 8; kt++) {
    v16h a = load_a_lds(abase + kt * 32, lane);
#pragma unroll
    for (int t = 0; t < 4; t++) {
      int nt = nq * 4 + t;
      aS[t] = wmma_f16(a, load_b_pk(W1s, kt, nt, lane), aS[t]);
      aT[t] = wmma_f16(a, load_b_pk(W1n, kt, nt, lane), aT[t]);
    }
  }
  int rbase = R + mt * 16 + ((lane >> 4) << 3);
#pragma unroll
  for (int t = 0; t < 4; t++) {
    int n = nq * 64 + t * 16 + (lane & 15);
    float bb = b1[n];
#pragma unroll
    for (int r = 0; r < 8; r++) {
      S[(size_t)(rbase + r) * DIM + n] = (_Float16)(aS[t][r] + bb);
      T[(size_t)(rbase + r) * DIM + n] = (_Float16)(aT[t][r]);
    }
  }
}

// ---------------- K2: fused message-avg + update MLP + LN ------------------
__global__ __launch_bounds__(256) void k_update(
    float* __restrict__ h32, _Float16* __restrict__ h16,
    const _Float16* __restrict__ S, const _Float16* __restrict__ T,
    const _Float16* __restrict__ W2, const float* __restrict__ b2,
    const _Float16* __restrict__ U1h, const _Float16* __restrict__ U1m,
    const float* __restrict__ ub1, const _Float16* __restrict__ U2,
    const float* __restrict__ ub2, const float* __restrict__ lng,
    const float* __restrict__ lnb, const int* __restrict__ flags) {
  if (!flags[0]) return;
  // buf1: Tx(36 rows) -> msgs M ; buf2: Sx -> hA ; buf3: G -> P
  // U32 (f32, 32xLDF) overlays buf1+buf2 for the residual+LN phase.
  __shared__ __align__(32) char smem[(36 + 32 + 32) * LDH * 2];
  _Float16* buf1 = (_Float16*)smem;
  _Float16* buf2 = buf1 + 36 * LDH;
  _Float16* buf3 = buf2 + 32 * LDH;
  float* U32 = (float*)smem;

  int tid = threadIdx.x, lane = tid & 31, wv = tid >> 5;
  int mt = wv >> 2, nq = wv & 3;
  int R = blockIdx.x * TILE_M;
  int b = R >> 13;       // batch (SEQ = 8192)
  int n0 = R & (SEQ - 1);

  // phase 1: stage S tile and T rows [n0-4, n0+32) (zero-fill causal pad)
  stage_rows(buf2, S + (size_t)R * DIM, tid);
  {
    int ch = tid & 7;
    for (int i = tid >> 3; i < 36; i += 32) {
      int nr = n0 - 4 + i;
      uint4* dst = (uint4*)(buf1 + i * LDH) + ch * 4;
      if (nr >= 0) {
        const uint4* src =
            (const uint4*)(T + ((size_t)(b << 13) + nr) * DIM) + ch * 4;
        dst[0] = src[0]; dst[1] = src[1]; dst[2] = src[2]; dst[3] = src[3];
      } else {
        uint4 z = {0, 0, 0, 0};
        dst[0] = z; dst[1] = z; dst[2] = z; dst[3] = z;
      }
    }
  }
  __syncthreads();

  // phase 2: G[r][c] = (1/5) sum_j gelu(S[r][c] + T[r+j][c])
  for (int idx = tid; idx < 32 * 256; idx += 256) {
    int r = idx >> 8, c = idx & 255;
    float s = (float)buf2[r * LDH + c];
    float acc = 0.f;
#pragma unroll
    for (int j = 0; j < 5; j++)
      acc += gelu_exact(s + (float)buf1[(r + j) * LDH + c]);
    buf3[r * LDH + c] = (_Float16)(acc * 0.2f);
  }
  __syncthreads();

  // phase 3: stage hA into buf2; GEMM1: M = G@W2 + b2 -> buf1
  stage_rows(buf2, h16 + (size_t)R * DIM, tid);
  {
    v8f acc[4] = {};
    const _Float16* abase = buf3 + mt * 16 * LDH;
#pragma unroll
    for (int kt = 0; kt < 8; kt++) {
      v16h a = load_a_lds(abase + kt * 32, lane);
#pragma unroll
      for (int t = 0; t < 4; t++)
        acc[t] = wmma_f16(a, load_b_pk(W2, kt, nq * 4 + t, lane), acc[t]);
    }
    int rb = mt * 16 + ((lane >> 4) << 3);
#pragma unroll
    for (int t = 0; t < 4; t++) {
      int n = nq * 64 + t * 16 + (lane & 15);
      float bb = b2[n];
#pragma unroll
      for (int r = 0; r < 8; r++)
        buf1[(rb + r) * LDH + n] = (_Float16)(acc[t][r] + bb);
    }
  }
  __syncthreads();

  // phase 4: GEMM2: P = gelu(hA@U1h + M@U1m + ub1) -> buf3
  {
    v8f acc[4] = {};
    const _Float16* a1 = buf2 + mt * 16 * LDH;
    const _Float16* a2 = buf1 + mt * 16 * LDH;
#pragma unroll
    for (int kt = 0; kt < 8; kt++) {
      v16h a = load_a_lds(a1 + kt * 32, lane);
#pragma unroll
      for (int t = 0; t < 4; t++)
        acc[t] = wmma_f16(a, load_b_pk(U1h, kt, nq * 4 + t, lane), acc[t]);
    }
#pragma unroll
    for (int kt = 0; kt < 8; kt++) {
      v16h a = load_a_lds(a2 + kt * 32, lane);
#pragma unroll
      for (int t = 0; t < 4; t++)
        acc[t] = wmma_f16(a, load_b_pk(U1m, kt, nq * 4 + t, lane), acc[t]);
    }
    int rb = mt * 16 + ((lane >> 4) << 3);
#pragma unroll
    for (int t = 0; t < 4; t++) {
      int n = nq * 64 + t * 16 + (lane & 15);
      float bb = ub1[n];
#pragma unroll
      for (int r = 0; r < 8; r++)
        buf3[(rb + r) * LDH + n] = (_Float16)gelu_exact(acc[t][r] + bb);
    }
  }
  __syncthreads();

  // phase 5: GEMM3: upd = P@U2 + ub2 ; U32 = upd + h (residual), overlay
  {
    v8f acc[4] = {};
    const _Float16* a1 = buf3 + mt * 16 * LDH;
#pragma unroll
    for (int kt = 0; kt < 8; kt++) {
      v16h a = load_a_lds(a1 + kt * 32, lane);
#pragma unroll
      for (int t = 0; t < 4; t++)
        acc[t] = wmma_f16(a, load_b_pk(U2, kt, nq * 4 + t, lane), acc[t]);
    }
    int rb = mt * 16 + ((lane >> 4) << 3);
#pragma unroll
    for (int t = 0; t < 4; t++) {
      int n = nq * 64 + t * 16 + (lane & 15);
      float bb = ub2[n];
#pragma unroll
      for (int r = 0; r < 8; r++) {
        int row = rb + r;
        U32[row * LDF + n] =
            acc[t][r] + bb + h32[(size_t)(R + row) * DIM + n];
      }
    }
  }
  __syncthreads();

  // phase 6: LayerNorm per row (8 lanes/row, shfl_xor reduce), write h back
  {
    int row = tid >> 3, sub = tid & 7;
    const float* x = U32 + row * LDF + sub * 32;
    float s = 0.f, ss = 0.f;
#pragma unroll
    for (int c = 0; c < 32; c++) {
      float v = x[c];
      s += v;
      ss += v * v;
    }
#pragma unroll
    for (int m = 1; m < 8; m <<= 1) {
      s += __shfl_xor(s, m, 32);
      ss += __shfl_xor(ss, m, 32);
    }
    float mu = s * (1.f / 256.f);
    float var = ss * (1.f / 256.f) - mu * mu;
    float rstd = rsqrtf(var + 1e-5f);
    int gcol = sub * 32;
    size_t gb = (size_t)(R + row) * DIM + gcol;
#pragma unroll
    for (int c = 0; c < 32; c++) {
      float v = (x[c] - mu) * rstd * lng[gcol + c] + lnb[gcol + c];
      h32[gb + c] = v;
      h16[gb + c] = (_Float16)v;
    }
  }
}

// --------------------- K3: halting prob (steps 2..4) -----------------------
__global__ __launch_bounds__(256) void k_halt_part(
    const float* __restrict__ h32, float* __restrict__ part,
    const int* __restrict__ flags) {
  if (!flags[0]) return;
  int t = threadIdx.x;
  size_t R = (size_t)blockIdx.x * TILE_M;
  float s = 0.f;
  for (int r = 0; r < TILE_M; r++) s += h32[(R + r) * DIM + t];
  part[(size_t)blockIdx.x * DIM + t] = s;
}

__global__ __launch_bounds__(256) void k_halt_reduce(
    const float* __restrict__ part, const float* __restrict__ hw,
    const float* __restrict__ hb, float* __restrict__ hp,
    const int* __restrict__ flags) {
  if (!flags[0]) return;
  __shared__ float red[256];
  int b = blockIdx.x, t = threadIdx.x;
  float s = 0.f;
  for (int g = 0; g < SEQ / TILE_M; g++)
    s += part[((size_t)b * (SEQ / TILE_M) + g) * DIM + t];
  red[t] = (s * (1.f / SEQ)) * hw[t];
  __syncthreads();
  for (int off = 128; off > 0; off >>= 1) {
    if (t < off) red[t] += red[t + off];
    __syncthreads();
  }
  if (t == 0) hp[b] = 1.f / (1.f + expf(-(red[0] + hb[0])));
}

// ------------------- K3c: scalar step logic (1 thread) ---------------------
__global__ void k_step_logic(int* flags, float* remaining, float* wout,
                             const float* hp, float* kl, float* d_kl_out,
                             int step, float p_geom) {
  if (threadIdx.x != 0 || blockIdx.x != 0) return;
  float hpe[4];
  for (int b = 0; b < 4; b++) {
    float v;
    if (step < 2) v = 0.f;
    else if (step == 5) v = 1.f;
    else v = hp[b];
    hpe[b] = v;
  }
  if (flags[0]) {
    float m = 0.25f * (hpe[0] + hpe[1] + hpe[2] + hpe[3]);
    kl[0] += p_geom * (logf(p_geom) - logf(m + 1e-8f));
    float rmax = 0.f;
    for (int b = 0; b < 4; b++) {
      wout[b] = remaining[b] * hpe[b];
      remaining[b] *= (1.f - hpe[b]);
      rmax = fmaxf(rmax, remaining[b]);
    }
    flags[0] = (rmax >= 0.01f) ? 1 : 0;
  } else {
    for (int b = 0; b < 4; b++) wout[b] = 0.f;
  }
  if (step == 5) d_kl_out[0] = kl[0] * (1.f / 6.f);
}

// ------------------------ K4: output accumulate ----------------------------
__global__ __launch_bounds__(256) void k_accum(float* __restrict__ out,
                                               const float* __restrict__ h32,
                                               const float* __restrict__ w) {
  size_t i = (size_t)blockIdx.x * 256 + threadIdx.x;
  out[i] += w[i >> 21] * h32[i];  // 8192*256 = 1<<21 elems per batch
}

// ---------------------------------------------------------------------------
extern "C" void kernel_launch(void* const* d_in, const int* in_sizes, int n_in,
                              void* d_out, int out_size, void* d_ws,
                              size_t ws_size, hipStream_t stream) {
  (void)in_sizes; (void)n_in; (void)out_size; (void)ws_size;
  const float* summaries = (const float*)d_in[0];
  const float* msg_w1 = (const float*)d_in[1];
  const float* msg_b1 = (const float*)d_in[2];
  const float* msg_w2 = (const float*)d_in[3];
  const float* msg_b2 = (const float*)d_in[4];
  const float* upd_w1 = (const float*)d_in[5];
  const float* upd_b1 = (const float*)d_in[6];
  const float* upd_w2 = (const float*)d_in[7];
  const float* upd_b2 = (const float*)d_in[8];
  const float* halt_w = (const float*)d_in[9];
  const float* halt_b = (const float*)d_in[10];
  const float* ln_g = (const float*)d_in[11];
  const float* ln_b = (const float*)d_in[12];
  float* out = (float*)d_out;

  char* ws = (char*)d_ws;
  int* flags = (int*)(ws + 0);
  float* remaining = (float*)(ws + 16);
  float* wstep = (float*)(ws + 32);
  float* hp = (float*)(ws + 48);
  float* kl = (float*)(ws + 64);
  _Float16* W1s = (_Float16*)(ws + 128);
  _Float16* W1n = W1s + 65536;
  _Float16* W2p = W1n + 65536;
  _Float16* U1h = W2p + 65536;
  _Float16* U1m = U1h + 65536;
  _Float16* U2p = U1m + 65536;
  const size_t ND = (size_t)ROWS_TOTAL * DIM;  // 8,388,608
  _Float16* h16 = (_Float16*)(ws + 786560);
  _Float16* Sb = h16 + ND;
  _Float16* Tb = Sb + ND;
  float* h32 = (float*)(Tb + ND);
  float* part = h32 + ND;  // [1024][256] f32 partials

  // one-time (per call) weight packing + state init
  k_pack<<<256, 256, 0, stream>>>(msg_w1, W1s, 0);
  k_pack<<<256, 256, 0, stream>>>(msg_w1, W1n, 256);
  k_pack<<<256, 256, 0, stream>>>(msg_w2, W2p, 0);
  k_pack<<<256, 256, 0, stream>>>(upd_w1, U1h, 0);
  k_pack<<<256, 256, 0, stream>>>(upd_w1, U1m, 256);
  k_pack<<<256, 256, 0, stream>>>(upd_w2, U2p, 0);
  k_init<<<32768, 256, 0, stream>>>(summaries, h32, h16, out, flags, remaining,
                                    kl);

  float pg = 0.2f;
  for (int step = 0; step < 6; step++) {
    k_proj<<<NUM_WG, 256, 0, stream>>>(h16, W1s, W1n, msg_b1, Sb, Tb, flags);
    k_update<<<NUM_WG, 256, 0, stream>>>(h32, h16, Sb, Tb, W2p, msg_b2, U1h,
                                         U1m, upd_b1, U2p, upd_b2, ln_g, ln_b,
                                         flags);
    if (step >= 2 && step < 5) {
      k_halt_part<<<NUM_WG, 256, 0, stream>>>(h32, part, flags);
      k_halt_reduce<<<4, 256, 0, stream>>>(part, halt_w, halt_b, hp, flags);
    }
    k_step_logic<<<1, 1, 0, stream>>>(flags, remaining, wstep, hp, kl,
                                      out + ND, step, pg);
    k_accum<<<32768, 256, 0, stream>>>(out, h32, wstep);
    pg *= 0.8f;
  }
}
